// MultiHeadAttention_37941741093588
// MI455X (gfx1250) — matrix operand order repaired and checked by reading, not compiled, not run on hardware
//
#include <hip/hip_runtime.h>
#include <hip/hip_bf16.h>

#define DM 512
#define DH 64
#define NH 8
#define BB 4
#define SS 2048
#define MROWS (BB * SS)          // 8192 token rows
#define NEG_INF_F (-1e9f)
#define LN_EPS 1e-5f

typedef _Float16 half_t;
typedef __attribute__((ext_vector_type(16))) _Float16 v16h;
typedef __attribute__((ext_vector_type(8)))  _Float16 v8h;
typedef __attribute__((ext_vector_type(8)))  float    v8f;
typedef __attribute__((ext_vector_type(4)))  unsigned int u32x4;
typedef __attribute__((ext_vector_type(8)))  int          i32x8;
typedef __attribute__((ext_vector_type(4)))  int          i32x4;

#if defined(__has_builtin)
#if __has_builtin(__builtin_amdgcn_tensor_load_to_lds) && \
    __has_builtin(__builtin_amdgcn_s_wait_tensorcnt)
#define USE_TDM 1
#endif
#endif
#ifndef USE_TDM
#define USE_TDM 0
#endif

// ---------------------------------------------------------------------------
// WMMA fragment loaders per CDNA5 VGPR layouts (CDNA5 ISA):
//  A (16x32 f16, MxK) row-major: lane l -> row m=l&15; idx0..7 -> K=hi*8+i,
//    idx8..15 -> K=16+hi*8+i (hi=l>>4).  Two contiguous 16B spans.
//  B (32x16 f16, KxN) stored NxK: lane l -> col n=l&15; same K order as A.
// LDS row strides are multiples of 8 halves -> all spans 16B aligned
// -> pairs of ds_load_b128.
// ---------------------------------------------------------------------------
__device__ __forceinline__ v16h frag_a(const half_t* base, int stride, int row, int hi) {
  const v8h lo = *(const v8h*)(base + row * stride + hi * 8);
  const v8h up = *(const v8h*)(base + row * stride + 16 + hi * 8);
  v16h r;
#pragma unroll
  for (int i = 0; i < 8; ++i) { r[i] = lo[i]; r[i + 8] = up[i]; }
  return r;
}

__device__ __forceinline__ v16h frag_b(const half_t* base, int stride, int n, int hi) {
  const v8h lo = *(const v8h*)(base + n * stride + hi * 8);
  const v8h up = *(const v8h*)(base + n * stride + 16 + hi * 8);
  v16h r;
#pragma unroll
  for (int i = 0; i < 8; ++i) { r[i] = lo[i]; r[i + 8] = up[i]; }
  return r;
}

#define WMMA_F32_F16(A, Bm, C) \
  __builtin_amdgcn_wmma_f32_16x16x32_f16(false, (A), false, (Bm), (short)0, (C), false, false)

#if USE_TDM
// ---------------------------------------------------------------------------
// Tensor Data Mover: 2D f16 tile load, global -> LDS, with hardware LDS row
// padding (pad_interval code 4 = 32 DWORDs = one 64-half row; pad_amount
// code 3 = 4 DWORDs = 8 halves) so rows land at stride 72 halves, matching
// the WMMA fragment loaders.  D# built per CDNA5 ISA.
// This toolchain exposes the 6-arg builtin:
//   (u32x4 g0, i32x8 g1, i32x4, i32x4, i32x8, i32 cpol)
// ---------------------------------------------------------------------------
__device__ __forceinline__ void tdm_load_tile_f16(
    unsigned lds_addr_bytes, unsigned long long gaddr_bytes,
    unsigned tile_d0, unsigned tile_d1, unsigned long long row_stride_elems) {
  u32x4 g0;
  g0[0] = 1u;                                          // count=1 (valid), user mode
  g0[1] = lds_addr_bytes;                              // lds_addr
  g0[2] = (unsigned)(gaddr_bytes & 0xFFFFFFFFull);     // global_addr[31:0]
  g0[3] = (unsigned)((gaddr_bytes >> 32) & 0x1FFFFFFull) | (2u << 30);  // addr[56:32]|type=2

  const unsigned td0 = 0x40000000u;                    // large tensor dims: no OOB clamp
  const unsigned td1 = 0x40000000u;
  i32x8 g1;
  g1[0] = (int)((1u << 16) |                           // data_size = 1 -> 2 bytes
                (1u << 20) |                           // pad_enable
                (4u << 22) |                           // pad_interval: 32 DWORDs
                (3u << 25));                           // pad_amount:   4 DWORDs
  g1[1] = (int)((td0 & 0xFFFFu) << 16);                // tensor_dim0[15:0] (atomic addr 0)
  g1[2] = (int)(((td0 >> 16) & 0xFFFFu) | ((td1 & 0xFFFFu) << 16));
  g1[3] = (int)(((td1 >> 16) & 0xFFFFu) | ((tile_d0 & 0xFFFFu) << 16));
  g1[4] = (int)(tile_d1 & 0xFFFFu);                    // tile_dim1 ; tile_dim2 = 0
  g1[5] = (int)(unsigned)(row_stride_elems & 0xFFFFFFFFull);   // dim0_stride[31:0]
  g1[6] = (int)(unsigned)((row_stride_elems >> 32) & 0xFFFFull);
  g1[7] = 0;

  const i32x4 z4 = {0, 0, 0, 0};                       // 2D: groups 2/3 unused
  const i32x8 z8 = {0, 0, 0, 0, 0, 0, 0, 0};
  __builtin_amdgcn_tensor_load_to_lds(g0, g1, z4, z4, z8, 0);
}
#endif

// ---------------------------------------------------------------------------
// Kernel 1: QKV projections. out(f16)[8192x512] = X(f32) @ W(f32) + bias.
// Block tile 128(M) x 128(N); 8 waves each own 32x64 (2x4 frags, 8 WMMAs).
// blockIdx.z selects q / k / v.
// ---------------------------------------------------------------------------
__global__ __launch_bounds__(256) void qkv_gemm_kernel(
    const float* __restrict__ xq, const float* __restrict__ xk, const float* __restrict__ xv,
    const float* __restrict__ Wq, const float* __restrict__ Wk, const float* __restrict__ Wv,
    const float* __restrict__ bq, const float* __restrict__ bk, const float* __restrict__ bv,
    half_t* __restrict__ qh, half_t* __restrict__ kh, half_t* __restrict__ vh) {
  const int which = blockIdx.z;
  const float* X    = which == 0 ? xq : (which == 1 ? xk : xv);
  const float* W    = which == 0 ? Wq : (which == 1 ? Wk : Wv);
  const float* bias = which == 0 ? bq : (which == 1 ? bk : bv);
  half_t* O         = which == 0 ? qh : (which == 1 ? kh : vh);

  const int n0 = blockIdx.x * 128;
  const int m0 = blockIdx.y * 128;

  __shared__ half_t As[128 * 40];  // M x K tile, stride 40 halves (80B, 16B-aligned rows)
  __shared__ half_t Bs[128 * 40];  // N x K tile (transposed W slice)

  const int tid  = threadIdx.x;
  const int wave = tid >> 5, lane = tid & 31;
  const int ln = lane & 15, hi = lane >> 4;
  const int wm = (wave & 3) * 32;   // wave M offset
  const int wn = (wave >> 2) * 64;  // wave N offset

  v8f acc[2][4];
#pragma unroll
  for (int i = 0; i < 2; ++i)
#pragma unroll
    for (int j = 0; j < 4; ++j) acc[i][j] = (v8f){0.f, 0.f, 0.f, 0.f, 0.f, 0.f, 0.f, 0.f};

  for (int k0 = 0; k0 < DM; k0 += 32) {
    // A: 128x32 fp32 -> f16
    {
      const int row = tid >> 1;
      const int cb  = (tid & 1) * 16;
      const float4* src = (const float4*)(X + (size_t)(m0 + row) * DM + k0 + cb);
#pragma unroll
      for (int v = 0; v < 4; ++v) {
        float4 f = src[v];
        half_t* dst = &As[row * 40 + cb + v * 4];
        dst[0] = (half_t)f.x; dst[1] = (half_t)f.y; dst[2] = (half_t)f.z; dst[3] = (half_t)f.w;
      }
      if (k0 + 32 < DM)
        __builtin_prefetch(X + (size_t)(m0 + row) * DM + k0 + 32 + cb, 0, 0);
    }
    // B: Bs[n][k] = W[k0+k][n0+n] (consecutive lanes -> consecutive n: coalesced)
    {
      const int n  = tid & 127;
      const int kg = tid >> 7;  // 0..1
#pragma unroll
      for (int kk = 0; kk < 16; ++kk) {
        const int k = kg * 16 + kk;
        Bs[n * 40 + k] = (half_t)W[(size_t)(k0 + k) * DM + n0 + n];
      }
      if (k0 + 32 < DM)
        __builtin_prefetch(W + (size_t)(k0 + 32 + kg * 16) * DM + n0 + n, 0, 0);
    }
    __syncthreads();

    v16h a0 = frag_a(As, 40, wm + ln, hi);
    v16h a1 = frag_a(As, 40, wm + 16 + ln, hi);
#pragma unroll
    for (int j = 0; j < 4; ++j) {
      v16h b = frag_b(Bs, 40, wn + j * 16 + ln, hi);
      acc[0][j] = WMMA_F32_F16(a0, b, acc[0][j]);
      acc[1][j] = WMMA_F32_F16(a1, b, acc[1][j]);
    }
    __syncthreads();
  }

  // Epilogue: C layout n = lane&15, m = r + 8*hi; add bias, store f16.
#pragma unroll
  for (int i = 0; i < 2; ++i)
#pragma unroll
    for (int j = 0; j < 4; ++j) {
      const int n_idx = n0 + wn + j * 16 + ln;
      const float bv_ = bias[n_idx];
#pragma unroll
      for (int r = 0; r < 8; ++r) {
        const int m = m0 + wm + i * 16 + 8 * hi + r;
        O[(size_t)m * DM + n_idx] = (half_t)(acc[i][j][r] + bv_);
      }
    }
}

// ---------------------------------------------------------------------------
// Kernel 2: flash-attention per (64-row q tile, head, batch).
// Q/K tiles staged by the Tensor Data Mover (with HW LDS padding); V
// transposed through registers; online softmax in fp32; WMMA everywhere.
// ---------------------------------------------------------------------------
__global__ __launch_bounds__(256) void attn_kernel(
    const half_t* __restrict__ qh, const half_t* __restrict__ kh,
    const half_t* __restrict__ vh, const unsigned char* __restrict__ mask,
    half_t* __restrict__ ctx) {
  const int qi = blockIdx.x;  // q tile (0..31)
  const int h  = blockIdx.y;  // head
  const int b  = blockIdx.z;  // batch

  __shared__ half_t Qs[64 * 72];   // Q tile, row-major M x d   (A layout)
  __shared__ half_t Ks[64 * 72];   // K tile, row-major kv x d  (B layout for QK^T)
  __shared__ half_t Vt[64 * 72];   // V tile transposed d x kv  (B layout for PV)
  __shared__ half_t Ps[64 * 72];   // probs f16, row-major q x kv (A layout)
  __shared__ float  Sc[64 * 65];   // fp32 scores
  __shared__ float  red_m[64 * 4];
  __shared__ float  red_s[64 * 4];
  __shared__ float  row_m[64], row_l[64], row_scale[64];

  const int tid  = threadIdx.x;
  const int wave = tid >> 5, lane = tid & 31;
  const int ln = lane & 15, hi = lane >> 4;
  const int wq = (wave & 3) * 16;   // this wave's q-row block
  const int wn = (wave >> 2) * 32;  // this wave's column block

  const size_t qbase = ((size_t)b * SS + (size_t)qi * 64) * DM + (size_t)h * DH;

  // Stage Q tile (64x64 f16, stride 72 in LDS).
#if USE_TDM
  if (wave == 0) {
    tdm_load_tile_f16((unsigned)(size_t)(void*)Qs,
                      (unsigned long long)(size_t)(const void*)(qh + qbase),
                      64u, 64u, (unsigned long long)DM);
    __builtin_amdgcn_s_wait_tensorcnt(0);
  }
#else
  {
    const int r = tid & 63, seg = tid >> 6;
    const v8h* src = (const v8h*)(qh + qbase + (size_t)r * DM + seg * 16);
    *(v8h*)&Qs[r * 72 + seg * 16]     = src[0];
    *(v8h*)&Qs[r * 72 + seg * 16 + 8] = src[1];
  }
#endif
  if (tid < 64) { row_m[tid] = -3.0e38f; row_l[tid] = 0.f; }

  v8f cacc[2];
  cacc[0] = (v8f){0.f, 0.f, 0.f, 0.f, 0.f, 0.f, 0.f, 0.f};
  cacc[1] = cacc[0];
  __syncthreads();

  for (int kv0 = 0; kv0 < SS; kv0 += 64) {
    const size_t kb0 = ((size_t)b * SS + (size_t)kv0) * DM + (size_t)h * DH;
    // Stage K via TDM; V via registers (transposed into Vt[d][kv]).
#if USE_TDM
    if (wave == 0)
      tdm_load_tile_f16((unsigned)(size_t)(void*)Ks,
                        (unsigned long long)(size_t)(const void*)(kh + kb0),
                        64u, 64u, (unsigned long long)DM);
#endif
    {
      const int r = tid & 63, seg = tid >> 6;
      const size_t kb = kb0 + (size_t)r * DM;
#if !USE_TDM
      const v8h* ks = (const v8h*)(kh + kb + seg * 16);
      *(v8h*)&Ks[r * 72 + seg * 16]     = ks[0];
      *(v8h*)&Ks[r * 72 + seg * 16 + 8] = ks[1];
#endif
      const v8h* vs = (const v8h*)(vh + kb + seg * 16);
      v8h v0 = vs[0], v1 = vs[1];
#pragma unroll
      for (int i = 0; i < 8; ++i) {
        Vt[(seg * 16 + i) * 72 + r]     = v0[i];
        Vt[(seg * 16 + 8 + i) * 72 + r] = v1[i];
      }
      if (kv0 + 64 < SS) {
        const size_t nb = kb0 + (size_t)64 * DM + (size_t)r * DM + seg * 16;
        __builtin_prefetch(kh + nb, 0, 0);
        __builtin_prefetch(vh + nb, 0, 0);
      }
    }
#if USE_TDM
    if (wave == 0) __builtin_amdgcn_s_wait_tensorcnt(0);
#endif
    __syncthreads();

    // scores(64x64) = Q @ K^T ; wave handles rows wq..+15, cols wn..+31.
    v8f s0 = (v8f){0.f, 0.f, 0.f, 0.f, 0.f, 0.f, 0.f, 0.f};
    v8f s1 = s0;
#pragma unroll
    for (int kk = 0; kk < 64; kk += 32) {
      v16h a  = frag_a(Qs + kk, 72, wq + ln, hi);
      v16h b0 = frag_b(Ks + kk, 72, wn + ln, hi);
      v16h b1 = frag_b(Ks + kk, 72, wn + 16 + ln, hi);
      s0 = WMMA_F32_F16(a, b0, s0);
      s1 = WMMA_F32_F16(a, b1, s1);
    }
#pragma unroll
    for (int r = 0; r < 8; ++r) {
      const int m = wq + 8 * hi + r;
      Sc[m * 65 + wn + ln]      = s0[r] * 0.125f;  // 1/sqrt(64)
      Sc[m * 65 + wn + 16 + ln] = s1[r] * 0.125f;
    }
    __syncthreads();

    // Phase A: mask + per-16-col partial max.
    {
      const int row = tid >> 2, seg = tid & 3;
      const unsigned char* mrow =
          mask + ((size_t)b * SS + (size_t)(qi * 64 + row)) * SS + kv0 + seg * 16;
      float mx = -3.0e38f;
#pragma unroll
      for (int i = 0; i < 16; ++i) {
        float s = Sc[row * 65 + seg * 16 + i];
        if (!mrow[i]) s = NEG_INF_F;
        Sc[row * 65 + seg * 16 + i] = s;
        mx = fmaxf(mx, s);
      }
      red_m[row * 4 + seg] = mx;
    }
    __syncthreads();
    // Phase B: running max + rescale factor.
    if (tid < 64) {
      const float mo = row_m[tid];
      const float mx = fmaxf(fmaxf(red_m[tid * 4 + 0], red_m[tid * 4 + 1]),
                             fmaxf(red_m[tid * 4 + 2], red_m[tid * 4 + 3]));
      const float mn = fmaxf(mo, mx);
      const float sc = __expf(mo - mn);
      row_m[tid] = mn;
      row_scale[tid] = sc;
      row_l[tid] *= sc;
    }
    __syncthreads();
    // Phase C: probs (fp32 exp -> f16) + partial sums.
    {
      const int row = tid >> 2, seg = tid & 3;
      const float mn = row_m[row];
      float sum = 0.f;
#pragma unroll
      for (int i = 0; i < 16; ++i) {
        const float p = __expf(Sc[row * 65 + seg * 16 + i] - mn);
        sum += p;
        Ps[row * 72 + seg * 16 + i] = (half_t)p;
      }
      red_s[row * 4 + seg] = sum;
    }
    __syncthreads();
    if (tid < 64)
      row_l[tid] += red_s[tid * 4 + 0] + red_s[tid * 4 + 1] +
                    red_s[tid * 4 + 2] + red_s[tid * 4 + 3];
    __syncthreads();

    // Rescale running ctx accumulators, then ctx += P @ V.
#pragma unroll
    for (int r = 0; r < 8; ++r) {
      const float sc = row_scale[wq + 8 * hi + r];
      cacc[0][r] *= sc;
      cacc[1][r] *= sc;
    }
#pragma unroll
    for (int kk = 0; kk < 64; kk += 32) {
      v16h a  = frag_a(Ps + kk, 72, wq + ln, hi);
      v16h b0 = frag_b(Vt + kk, 72, wn + ln, hi);
      v16h b1 = frag_b(Vt + kk, 72, wn + 16 + ln, hi);
      cacc[0] = WMMA_F32_F16(a, b0, cacc[0]);
      cacc[1] = WMMA_F32_F16(a, b1, cacc[1]);
    }
    __syncthreads();
  }

  // Normalize by row sum; emit ctx f16 in [B*S, H*64] layout for out-proj.
#pragma unroll
  for (int j = 0; j < 2; ++j)
#pragma unroll
    for (int r = 0; r < 8; ++r) {
      const int m = wq + 8 * hi + r;
      const int d = wn + j * 16 + ln;
      const float val = cacc[j][r] / row_l[m];
      ctx[((size_t)b * SS + (size_t)(qi * 64 + m)) * DM + (size_t)h * DH + d] = (half_t)val;
    }
}

// ---------------------------------------------------------------------------
// Kernel 3: y(f32) = ctx(f16) @ Wo(f32->f16) + bo + x_q  (pre-LayerNorm)
// Block tile 128x128, wave tile 32x64, as kernel 1.
// ---------------------------------------------------------------------------
__global__ __launch_bounds__(256) void out_gemm_kernel(
    const half_t* __restrict__ ctx, const float* __restrict__ Wo,
    const float* __restrict__ bo, const float* __restrict__ xq,
    float* __restrict__ y) {
  const int n0 = blockIdx.x * 128;
  const int m0 = blockIdx.y * 128;

  __shared__ half_t As[128 * 40];
  __shared__ half_t Bs[128 * 40];

  const int tid  = threadIdx.x;
  const int wave = tid >> 5, lane = tid & 31;
  const int ln = lane & 15, hi = lane >> 4;
  const int wm = (wave & 3) * 32;
  const int wn = (wave >> 2) * 64;

  v8f acc[2][4];
#pragma unroll
  for (int i = 0; i < 2; ++i)
#pragma unroll
    for (int j = 0; j < 4; ++j) acc[i][j] = (v8f){0.f, 0.f, 0.f, 0.f, 0.f, 0.f, 0.f, 0.f};

  for (int k0 = 0; k0 < DM; k0 += 32) {
    {
      const int row = tid >> 1;
      const int cb  = (tid & 1) * 16;
      const v8h* src = (const v8h*)(ctx + (size_t)(m0 + row) * DM + k0 + cb);
      *(v8h*)&As[row * 40 + cb]     = src[0];
      *(v8h*)&As[row * 40 + cb + 8] = src[1];
    }
    {
      const int n  = tid & 127;
      const int kg = tid >> 7;
#pragma unroll
      for (int kk = 0; kk < 16; ++kk) {
        const int k = kg * 16 + kk;
        Bs[n * 40 + k] = (half_t)Wo[(size_t)(k0 + k) * DM + n0 + n];
      }
      if (k0 + 32 < DM)
        __builtin_prefetch(Wo + (size_t)(k0 + 32 + kg * 16) * DM + n0 + n, 0, 0);
    }
    __syncthreads();

    v16h a0 = frag_a(As, 40, wm + ln, hi);
    v16h a1 = frag_a(As, 40, wm + 16 + ln, hi);
#pragma unroll
    for (int j = 0; j < 4; ++j) {
      v16h b = frag_b(Bs, 40, wn + j * 16 + ln, hi);
      acc[0][j] = WMMA_F32_F16(a0, b, acc[0][j]);
      acc[1][j] = WMMA_F32_F16(a1, b, acc[1][j]);
    }
    __syncthreads();
  }

#pragma unroll
  for (int i = 0; i < 2; ++i)
#pragma unroll
    for (int j = 0; j < 4; ++j) {
      const int n_idx = n0 + wn + j * 16 + ln;
      const float bv_ = bo[n_idx];
#pragma unroll
      for (int r = 0; r < 8; ++r) {
        const int m = m0 + wm + i * 16 + 8 * hi + r;
        y[(size_t)m * DM + n_idx] = acc[i][j][r] + bv_ + xq[(size_t)m * DM + n_idx];
      }
    }
}

// ---------------------------------------------------------------------------
// Kernel 4: LayerNorm over last dim (512), one block per row.
// ---------------------------------------------------------------------------
__global__ __launch_bounds__(256) void ln_kernel(
    const float* __restrict__ y, const float* __restrict__ gamma,
    const float* __restrict__ beta, float* __restrict__ out) {
  __shared__ float ssum[256];
  __shared__ float ssq[256];
  const int row = blockIdx.x;
  const int tid = threadIdx.x;

  const float a = y[(size_t)row * DM + tid];
  const float c = y[(size_t)row * DM + tid + 256];
  ssum[tid] = a + c;
  ssq[tid]  = a * a + c * c;
  __syncthreads();
#pragma unroll
  for (int off = 128; off > 0; off >>= 1) {
    if (tid < off) { ssum[tid] += ssum[tid + off]; ssq[tid] += ssq[tid + off]; }
    __syncthreads();
  }
  const float mu  = ssum[0] * (1.f / DM);
  const float var = ssq[0] * (1.f / DM) - mu * mu;
  const float rs  = rsqrtf(var + LN_EPS);
  out[(size_t)row * DM + tid]       = (a - mu) * rs * gamma[tid] + beta[tid];
  out[(size_t)row * DM + tid + 256] = (c - mu) * rs * gamma[tid + 256] + beta[tid + 256];
}

// ---------------------------------------------------------------------------
// Launch. Inputs (setup_inputs order):
//  0 x_q 1 x_k 2 x_v 3 mask(bool) 4 Wq 5 bq 6 Wk 7 bk 8 Wv 9 bv
//  10 Wo 11 bo 12 gamma 13 beta
// Workspace: q/k/v/ctx f16 (4 x 8 MiB) + y fp32 (16 MiB) = 48 MiB.
// ---------------------------------------------------------------------------
extern "C" void kernel_launch(void* const* d_in, const int* in_sizes, int n_in,
                              void* d_out, int out_size, void* d_ws, size_t ws_size,
                              hipStream_t stream) {
  const float* x_q  = (const float*)d_in[0];
  const float* x_k  = (const float*)d_in[1];
  const float* x_v  = (const float*)d_in[2];
  const unsigned char* mask = (const unsigned char*)d_in[3];
  const float* Wq = (const float*)d_in[4];
  const float* bq = (const float*)d_in[5];
  const float* Wk = (const float*)d_in[6];
  const float* bk = (const float*)d_in[7];
  const float* Wv = (const float*)d_in[8];
  const float* bv = (const float*)d_in[9];
  const float* Wo = (const float*)d_in[10];
  const float* bo = (const float*)d_in[11];
  const float* gamma = (const float*)d_in[12];
  const float* beta  = (const float*)d_in[13];
  float* out = (float*)d_out;

  const size_t half_buf = (size_t)MROWS * DM * sizeof(half_t);  // 8 MiB
  char* ws = (char*)d_ws;
  half_t* qhb = (half_t*)(ws);
  half_t* khb = (half_t*)(ws + half_buf);
  half_t* vhb = (half_t*)(ws + 2 * half_buf);
  half_t* ctx = (half_t*)(ws + 3 * half_buf);
  float*  yb  = (float*)(ws + 4 * half_buf);                    // 16 MiB

  qkv_gemm_kernel<<<dim3(DM / 128, MROWS / 128, 3), 256, 0, stream>>>(
      x_q, x_k, x_v, Wq, Wk, Wv, bq, bk, bv, qhb, khb, vhb);

  attn_kernel<<<dim3(SS / 64, NH, BB), 256, 0, stream>>>(qhb, khb, vhb, mask, ctx);

  out_gemm_kernel<<<dim3(DM / 128, MROWS / 128), 256, 0, stream>>>(ctx, Wo, bo, x_q, yb);

  ln_kernel<<<dim3(MROWS), 256, 0, stream>>>(yb, gamma, beta, out);
}
